// Encoder_70093866271440
// MI455X (gfx1250) — matrix operand order repaired and checked by reading, not compiled, not run on hardware
//
#include <hip/hip_runtime.h>

// ---------------------------------------------------------------------------
// Types & helpers
// ---------------------------------------------------------------------------
typedef __attribute__((ext_vector_type(8)))  float  v8f;
typedef __attribute__((ext_vector_type(16))) __bf16 v16bf;
typedef unsigned short u16;

union BF16x16 { v16bf v; uint4 q[2]; };

static __device__ inline u16 f2bf(float f) {
    unsigned u = __float_as_uint(f);
    unsigned r = u + 0x7FFFu + ((u >> 16) & 1u);   // round-to-nearest-even
    return (u16)(r >> 16);
}
static __device__ inline float sigf(float x) { return 1.0f / (1.0f + __expf(-x)); }

#define H 512
#define G4H 2048

// ---------------------------------------------------------------------------
// fp32 -> bf16 convert ; bias combine (bih + bhh)
// ---------------------------------------------------------------------------
__global__ void k_f2bf(const float* __restrict__ src, u16* __restrict__ dst, int n) {
    int i = blockIdx.x * blockDim.x + threadIdx.x;
    if (i < n) dst[i] = f2bf(src[i]);
}
__global__ void k_bias2(const float* __restrict__ a, const float* __restrict__ b,
                        float* __restrict__ out, int n) {
    int i = blockIdx.x * blockDim.x + threadIdx.x;
    if (i < n) out[i] = a[i] + b[i];
}

// ---------------------------------------------------------------------------
// Embedding gather -> bf16 x buffer, layout [T, Mp, 512], zero pad rows
// ---------------------------------------------------------------------------
__global__ void k_gather(const float* __restrict__ emb, const int* __restrict__ tok,
                         u16* __restrict__ x, int Mreal, int Mp, int T) {
    int id = blockIdx.x * blockDim.x + threadIdx.x;
    int total = T * Mp * 512;
    if (id >= total) return;
    int k = id & 511;
    int m = (id >> 9) % Mp;
    int t = id / (512 * Mp);
    float v = 0.0f;
    if (m < Mreal) v = emb[(size_t)tok[m * T + t] * 512 + k];
    x[id] = f2bf(v);
}

// ---------------------------------------------------------------------------
// Generic WMMA GEMM: C[m,n] = sum_k A[m,k]*B[n,k] (+ bias[n])
// A: bf16 [M,K] row-major.  B: bf16 [N,K] row-major (i.e. B^T).  C: fp32 [M,N].
// M,N multiples of 16; K multiple of 32.  One 16x16 tile per wave.
// ---------------------------------------------------------------------------
__global__ __launch_bounds__(256) void k_gemm(const u16* __restrict__ A,
                                              const u16* __restrict__ B,
                                              const float* __restrict__ bias,
                                              float* __restrict__ C,
                                              int M, int N, int K) {
    int Nt = N >> 4, Mt = M >> 4;
    int tile = blockIdx.x * 8 + (threadIdx.x >> 5);
    if (tile >= Mt * Nt) return;
    int mt = tile / Nt, nt = tile % Nt;
    int lane = threadIdx.x & 31;
    int lhalf = (lane >> 4) & 1;
    int l15 = lane & 15;

    v8f acc = {0.f, 0.f, 0.f, 0.f, 0.f, 0.f, 0.f, 0.f};
    const u16* ap = A + (size_t)(mt * 16 + l15) * K + lhalf * 8;   // A frag: K octets {0-7,16-23} / {8-15,24-31}
    const u16* bp = B + (size_t)(nt * 16 + l15) * K + lhalf * 8;   // B frag: K octets {0-7,16-23} / {8-15,24-31}

    for (int k0 = 0; k0 < K; k0 += 32) {
        BF16x16 a, b;
        a.q[0] = *(const uint4*)(ap + k0);
        a.q[1] = *(const uint4*)(ap + k0 + 16);
        b.q[0] = *(const uint4*)(bp + k0);
        b.q[1] = *(const uint4*)(bp + k0 + 16);
        acc = __builtin_amdgcn_wmma_f32_16x16x32_bf16(false, a.v, false, b.v,
                                                      (short)0, acc, false, false);
    }
    int col = nt * 16 + l15;
    float bv = bias ? bias[col] : 0.0f;
#pragma unroll
    for (int j = 0; j < 8; ++j) {
        int row = mt * 16 + lhalf * 8 + j;   // C layout: VGPR j -> M=j / j+8
        C[(size_t)row * N + col] = acc[j] + bv;
    }
}

// ---------------------------------------------------------------------------
// Persistent LSTM recurrence: one workgroup (32 waves), h resident in LDS,
// c resident in VGPRs.  gates = xw[t] + h @ Whh^T ; per-wave tiles cover all
// four gates of one 16-column block so the nonlinearity stays in registers.
// xw: fp32 [T,Mp,2048] (includes bih+bhh).  Whh: bf16 [2048,512].
// ---------------------------------------------------------------------------
__global__ __launch_bounds__(1024) void k_lstm_steps(const float* __restrict__ xw,
                                                     const u16* __restrict__ Whh,
                                                     u16* __restrict__ hseq,
                                                     float* __restrict__ hT,
                                                     float* __restrict__ cT,
                                                     int Mp, int T) {
    __shared__ __align__(16) u16 hlds[32 * 512];
    const int tid   = threadIdx.x;
    const int wave  = tid >> 5;
    const int lane  = tid & 31;
    const int lhalf = (lane >> 4) & 1;
    const int l15   = lane & 15;
    const int Mt    = Mp >> 4;
    const int npp   = (Mt == 2) ? 2 : 1;     // pairs per wave (uniform)
    const int jb    = wave;                  // output column block 0..31
    const int col   = jb * 16 + l15;

    for (int i = tid; i < Mp * 512; i += 1024) hlds[i] = 0;
    __syncthreads();

    v8f zero = {0.f, 0.f, 0.f, 0.f, 0.f, 0.f, 0.f, 0.f};
    v8f creg[2]; creg[0] = zero; creg[1] = zero;
    v8f hnew[2]; hnew[0] = zero; hnew[1] = zero;

    for (int t = 0; t < T; ++t) {
#pragma unroll
        for (int pp = 0; pp < 2; ++pp) {
            if (pp < npp) {
                const int mt = pp;           // M tile
                v8f acc[4];
#pragma unroll
                for (int g = 0; g < 4; ++g) {
#pragma unroll
                    for (int j = 0; j < 8; ++j) {
                        int row = mt * 16 + lhalf * 8 + j;
                        acc[g][j] = xw[(size_t)(t * Mp + row) * G4H + g * 512 + col];
                    }
                }
                const int arow = mt * 16 + l15;
                for (int k0 = 0; k0 < 512; k0 += 32) {
                    BF16x16 a;
                    int abase = arow * 512 + k0 + lhalf * 8;
                    a.q[0] = *(const uint4*)&hlds[abase];
                    a.q[1] = *(const uint4*)&hlds[abase + 16];
#pragma unroll
                    for (int g = 0; g < 4; ++g) {
                        BF16x16 b;
                        const u16* bp = Whh + (size_t)(g * 512 + jb * 16 + l15) * 512
                                            + k0 + lhalf * 8;
                        b.q[0] = *(const uint4*)bp;
                        b.q[1] = *(const uint4*)(bp + 16);
                        acc[g] = __builtin_amdgcn_wmma_f32_16x16x32_bf16(
                            false, a.v, false, b.v, (short)0, acc[g], false, false);
                    }
                }
#pragma unroll
                for (int j = 0; j < 8; ++j) {
                    float ig = sigf(acc[0][j]);
                    float fg = sigf(acc[1][j]);
                    float gg = tanhf(acc[2][j]);
                    float og = sigf(acc[3][j]);
                    float cc = fg * creg[pp][j] + ig * gg;
                    creg[pp][j] = cc;
                    hnew[pp][j] = og * tanhf(cc);
                }
            }
        }
        __syncthreads();                      // everyone done reading h(t-1)
#pragma unroll
        for (int pp = 0; pp < 2; ++pp) {
            if (pp < npp) {
                const int mt = pp;
#pragma unroll
                for (int j = 0; j < 8; ++j) {
                    int row = mt * 16 + lhalf * 8 + j;
                    u16 hb = f2bf(hnew[pp][j]);
                    hlds[row * 512 + col] = hb;
                    hseq[(size_t)(t * Mp + row) * 512 + col] = hb;
                }
            }
        }
        __syncthreads();                      // h(t) visible to all waves
    }
#pragma unroll
    for (int pp = 0; pp < 2; ++pp) {
        if (pp < npp) {
            const int mt = pp;
#pragma unroll
            for (int j = 0; j < 8; ++j) {
                int row = mt * 16 + lhalf * 8 + j;
                hT[(size_t)row * 512 + col] = hnew[pp][j];
                cT[(size_t)row * 512 + col] = creg[pp][j];
            }
        }
    }
}

// ---------------------------------------------------------------------------
// TreeLSTM (64 trees, binary heap of 63 nodes, levels lvl=0..5 bottom-up)
// ---------------------------------------------------------------------------
static __device__ inline void tree_feats(const float* __restrict__ fo,
                                         const float* __restrict__ fc,
                                         int tree, int node, float& f0, float& f1) {
    const float* f = (tree < 32) ? (fo + ((size_t)tree * 63 + node) * 2)
                                 : (fc + ((size_t)(tree - 32) * 63 + node) * 2);
    f0 = f[0]; f1 = f[1];
}

// lvl 0: leaves 31..62 : h_sum = 0
__global__ void k_tree_lvl0(const float* __restrict__ fo, const float* __restrict__ fc,
                            const float* __restrict__ Wiou, const float* __restrict__ biou,
                            float* __restrict__ h, u16* __restrict__ hbf,
                            float* __restrict__ c) {
    int id = blockIdx.x * blockDim.x + threadIdx.x;
    if (id >= 64 * 32 * 512) return;
    int j = id & 511;
    int ni = (id >> 9) & 31;
    int tree = id >> 14;
    int node = 31 + ni;
    float f0, f1; tree_feats(fo, fc, tree, node, f0, f1);
    float ip = f0 * Wiou[2 * j]           + f1 * Wiou[2 * j + 1]           + biou[j];
    float op = f0 * Wiou[2 * (512 + j)]   + f1 * Wiou[2 * (512 + j) + 1]   + biou[512 + j];
    float up = f0 * Wiou[2 * (1024 + j)]  + f1 * Wiou[2 * (1024 + j) + 1]  + biou[1024 + j];
    float cc = sigf(ip) * tanhf(up);
    float hh = sigf(op) * tanhf(cc);
    size_t idx = ((size_t)tree * 63 + node) * 512 + j;
    h[idx] = hh; c[idx] = cc; hbf[idx] = f2bf(hh);
}

// gather children sums (bf16) + per-child h (bf16) for level's GEMMs
__global__ void k_tree_gather(const float* __restrict__ h, const u16* __restrict__ hbf,
                              u16* __restrict__ hsum, u16* __restrict__ hch, int P) {
    int id = blockIdx.x * blockDim.x + threadIdx.x;
    int R = 64 * P;
    if (id >= R * 512) return;
    int j = id & 511;
    int r = id >> 9;
    int tree = r / P, pi = r % P;
    int p = P - 1 + pi;
    size_t tb = (size_t)tree * 63;
    size_t c1 = (tb + 2 * p + 1) * 512 + j;
    size_t c2 = (tb + 2 * p + 2) * 512 + j;
    hsum[(size_t)r * 512 + j] = f2bf(h[c1] + h[c2]);
    hch[(size_t)(2 * r) * 512 + j]     = hbf[c1];
    hch[(size_t)(2 * r + 1) * 512 + j] = hbf[c2];
}

// level update after the two GEMMs
__global__ void k_tree_update(const float* __restrict__ fo, const float* __restrict__ fc,
                              const float* __restrict__ Wiou, const float* __restrict__ biou,
                              const float* __restrict__ Wf,   const float* __restrict__ bf,
                              const float* __restrict__ iou,  const float* __restrict__ fpre,
                              float* __restrict__ h, u16* __restrict__ hbf,
                              float* __restrict__ c, int P) {
    int id = blockIdx.x * blockDim.x + threadIdx.x;
    int R = 64 * P;
    if (id >= R * 512) return;
    int j = id & 511;
    int r = id >> 9;
    int tree = r / P, pi = r % P;
    int p = P - 1 + pi;
    float f0, f1; tree_feats(fo, fc, tree, p, f0, f1);
    size_t ib = (size_t)r * 1536;
    float ip = f0 * Wiou[2 * j]          + f1 * Wiou[2 * j + 1]          + biou[j]        + iou[ib + j];
    float op = f0 * Wiou[2 * (512 + j)]  + f1 * Wiou[2 * (512 + j) + 1]  + biou[512 + j]  + iou[ib + 512 + j];
    float up = f0 * Wiou[2 * (1024 + j)] + f1 * Wiou[2 * (1024 + j) + 1] + biou[1024 + j] + iou[ib + 1024 + j];
    float wf = f0 * Wf[2 * j] + f1 * Wf[2 * j + 1] + bf[j];
    float fg1 = sigf(wf + fpre[(size_t)(2 * r) * 512 + j]);
    float fg2 = sigf(wf + fpre[(size_t)(2 * r + 1) * 512 + j]);
    size_t tb = (size_t)tree * 63;
    float c1 = c[(tb + 2 * p + 1) * 512 + j];
    float c2 = c[(tb + 2 * p + 2) * 512 + j];
    float cc = sigf(ip) * tanhf(up) + fg1 * c1 + fg2 * c2;
    float hh = sigf(op) * tanhf(cc);
    size_t idx = (tb + p) * 512 + j;
    h[idx] = hh; c[idx] = cc; hbf[idx] = f2bf(hh);
}

// ---------------------------------------------------------------------------
// Dense head (tiny fp32)
// ---------------------------------------------------------------------------
__global__ void k_dense_ast(const float* __restrict__ trh, const float* __restrict__ Wdh,
                            const float* __restrict__ bdh, float* __restrict__ h_ast) {
    int id = blockIdx.x * blockDim.x + threadIdx.x;
    if (id >= 32 * 512) return;
    int j = id & 511, bc = id >> 9;
    const float* ho = trh + (size_t)bc * 63 * 512;          // h_root old (node 0)
    const float* hc = trh + (size_t)(32 + bc) * 63 * 512;   // h_root cur
    float acc = bdh[j];
    const float* w = Wdh + (size_t)j * 1024;
    for (int k = 0; k < 512; ++k)
        acc += ho[k] * w[k] + hc[k] * w[512 + k];
    h_ast[bc * 512 + j] = acc;
}

__global__ void k_dense_am(const float* __restrict__ h_ast, const float* __restrict__ Wmh,
                           const float* __restrict__ bmh, float* __restrict__ h_am) {
    int bc = threadIdx.x;
    if (bc >= 32) return;
    float acc = bmh[0];
    for (int k = 0; k < 512; ++k) acc += h_ast[bc * 512 + k] * Wmh[k];
    h_am[bc] = acc;
}

__global__ void k_mrg(const float* __restrict__ hT_sc, const float* __restrict__ cT_sc,
                      const float* __restrict__ hT_cm, const float* __restrict__ cT_cm,
                      const float* __restrict__ h_am,
                      const float* __restrict__ Wgh, const float* __restrict__ bgh,
                      const float* __restrict__ Wgc, const float* __restrict__ bgc,
                      float* __restrict__ h_mrg, float* __restrict__ c_mrg) {
    int id = threadIdx.x;
    if (id >= 64) return;
    int cc = id & 3, l = (id >> 2) & 1, b = id >> 3;
    int bc = b * 4 + cc;
    size_t base = (size_t)l * 32 * 512 + (size_t)bc * 512;
    float ah = bgh[0] + h_am[bc] * Wgh[1024];
    float ac = bgc[0] + h_am[bc] * Wgc[1024];
    for (int k = 0; k < 512; ++k) {
        ah += hT_sc[base + k] * Wgh[k] + hT_cm[base + k] * Wgh[512 + k];
        ac += cT_sc[base + k] * Wgc[k] + cT_cm[base + k] * Wgc[512 + k];
    }
    int o = (b * 2 + l) * 4 + cc;     // [B, L, C]
    h_mrg[o] = ah; c_mrg[o] = ac;
}

__global__ void k_final(const float* __restrict__ h_mrg, const float* __restrict__ c_mrg,
                        const float* __restrict__ hT_it, const float* __restrict__ cT_it,
                        const float* __restrict__ Wfh, const float* __restrict__ bfh,
                        const float* __restrict__ Wfc, const float* __restrict__ bfc,
                        float* __restrict__ out) {
    int id = blockIdx.x * blockDim.x + threadIdx.x;
    if (id >= 16384) return;
    int j  = id & 511;
    int b  = (id >> 9) & 7;
    int l  = (id >> 12) & 1;
    int hc = id >> 13;
    const float* W    = hc ? Wfc : Wfh;
    const float* bias = hc ? bfc : bfh;
    const float* mrg  = hc ? c_mrg : h_mrg;
    const float* it   = hc ? cT_it : hT_it;
    const float* w = W + (size_t)j * 516;
    float acc = bias[j];
    for (int k = 0; k < 4; ++k) acc += mrg[(b * 2 + l) * 4 + k] * w[k];
    const float* itv = it + (size_t)l * 16 * 512 + (size_t)b * 512;  // it states padded Mp=16
    for (int k = 0; k < 512; ++k) acc += itv[k] * w[4 + k];
    out[(size_t)hc * 8192 + (size_t)l * 4096 + (size_t)b * 512 + j] = acc;
}

// ---------------------------------------------------------------------------
// Host orchestration
// ---------------------------------------------------------------------------
extern "C" void kernel_launch(void* const* d_in, const int* in_sizes, int n_in,
                              void* d_out, int out_size, void* d_ws, size_t ws_size,
                              hipStream_t stream) {
    (void)in_sizes; (void)n_in; (void)out_size; (void)ws_size;
    // input indices per reference signature
    const float* emb[3]    = { (const float*)d_in[1], (const float*)d_in[0], (const float*)d_in[2] }; // sc, cm, it
    const int*   toks[3]   = { (const int*)d_in[39],  (const int*)d_in[40],  (const int*)d_in[41] };
    const float* Wih_f[3]  = { (const float*)d_in[7], (const float*)d_in[3], (const float*)d_in[11] };
    const float* Whh_f[3]  = { (const float*)d_in[8], (const float*)d_in[4], (const float*)d_in[12] };
    const float* bih_f[3]  = { (const float*)d_in[9], (const float*)d_in[5], (const float*)d_in[13] };
    const float* bhh_f[3]  = { (const float*)d_in[10],(const float*)d_in[6], (const float*)d_in[14] };
    const float* tW_iou = (const float*)d_in[15];
    const float* tb_iou = (const float*)d_in[16];
    const float* tU_iou = (const float*)d_in[17];
    const float* tW_f   = (const float*)d_in[18];
    const float* tb_f   = (const float*)d_in[19];
    const float* tU_f   = (const float*)d_in[20];
    const float* Wdh = (const float*)d_in[21]; const float* bdh = (const float*)d_in[22];
    const float* Wmh = (const float*)d_in[25]; const float* bmh = (const float*)d_in[26];
    const float* Wgh = (const float*)d_in[29]; const float* bgh = (const float*)d_in[30];
    const float* Wgc = (const float*)d_in[31]; const float* bgc = (const float*)d_in[32];
    const float* Wfh = (const float*)d_in[33]; const float* bfh = (const float*)d_in[34];
    const float* Wfc = (const float*)d_in[35]; const float* bfc = (const float*)d_in[36];
    const float* feat_old = (const float*)d_in[37];
    const float* feat_cur = (const float*)d_in[38];

    const int Mp[3]    = { 32, 32, 16 };   // padded step batch (sc, cm, it)
    const int Mreal[3] = { 32, 32, 8 };
    const int T = 16;

    // ---- workspace layout ----
    char* ws = (char*)d_ws;
    size_t off = 0;
    auto alloc = [&](size_t bytes) -> char* {
        char* p = ws + off;
        off = (off + bytes + 255) & ~(size_t)255;
        return p;
    };
    u16*   wih_bf[3]; u16* whh_bf[3]; float* biasc[3];
    float* hT[3]; float* cT[3];
    for (int i = 0; i < 3; ++i) {
        wih_bf[i] = (u16*)alloc((size_t)2 * G4H * H * 2);
        whh_bf[i] = (u16*)alloc((size_t)2 * G4H * H * 2);
        biasc[i]  = (float*)alloc((size_t)2 * G4H * 4);
        hT[i]     = (float*)alloc((size_t)2 * Mp[i] * H * 4);
        cT[i]     = (float*)alloc((size_t)2 * Mp[i] * H * 4);
    }
    u16*   tUiou_bf = (u16*)alloc((size_t)1536 * 512 * 2);
    u16*   tUf_bf   = (u16*)alloc((size_t)512 * 512 * 2);
    u16*   xbuf     = (u16*)alloc((size_t)512 * 512 * 2);       // shared across LSTMs
    float* xwbuf    = (float*)alloc((size_t)512 * 2048 * 4);    // shared
    u16*   hseq     = (u16*)alloc((size_t)512 * 512 * 2);       // shared
    float* tr_h     = (float*)alloc((size_t)64 * 63 * 512 * 4);
    float* tr_c     = (float*)alloc((size_t)64 * 63 * 512 * 4);
    u16*   tr_hbf   = (u16*)alloc((size_t)64 * 63 * 512 * 2);
    u16*   hsum_bf  = (u16*)alloc((size_t)1024 * 512 * 2);
    u16*   hch_bf   = (u16*)alloc((size_t)2048 * 512 * 2);
    float* iou      = (float*)alloc((size_t)1024 * 1536 * 4);
    float* fpre     = (float*)alloc((size_t)2048 * 512 * 4);
    float* h_ast    = (float*)alloc((size_t)32 * 512 * 4);
    float* h_am     = (float*)alloc(32 * 4);
    float* h_mrg    = (float*)alloc(64 * 4);
    float* c_mrg    = (float*)alloc(64 * 4);

    auto gemm = [&](const u16* A, const u16* B, const float* bias, float* C,
                    int M, int N, int K) {
        int tiles = (M / 16) * (N / 16);
        k_gemm<<<(tiles + 7) / 8, 256, 0, stream>>>(A, B, bias, C, M, N, K);
    };

    // ---- weight conversion ----
    const int nw = 2 * G4H * H;
    for (int i = 0; i < 3; ++i) {
        k_f2bf<<<(nw + 255) / 256, 256, 0, stream>>>(Wih_f[i], wih_bf[i], nw);
        k_f2bf<<<(nw + 255) / 256, 256, 0, stream>>>(Whh_f[i], whh_bf[i], nw);
        k_bias2<<<(2 * G4H + 255) / 256, 256, 0, stream>>>(bih_f[i], bhh_f[i], biasc[i], 2 * G4H);
    }
    k_f2bf<<<(1536 * 512 + 255) / 256, 256, 0, stream>>>(tU_iou, tUiou_bf, 1536 * 512);
    k_f2bf<<<(512 * 512 + 255) / 256, 256, 0, stream>>>(tU_f, tUf_bf, 512 * 512);

    // ---- three LSTMs ----
    for (int i = 0; i < 3; ++i) {
        int mp = Mp[i], rows = T * mp;
        int nthr = rows * 512;
        k_gather<<<(nthr + 255) / 256, 256, 0, stream>>>(emb[i], toks[i], xbuf, Mreal[i], mp, T);
        // layer 0
        gemm(xbuf, wih_bf[i], biasc[i], xwbuf, rows, G4H, H);
        k_lstm_steps<<<1, 1024, 0, stream>>>(xwbuf, whh_bf[i], hseq,
                                             hT[i], cT[i], mp, T);
        // layer 1
        gemm(hseq, wih_bf[i] + (size_t)G4H * H, biasc[i] + G4H, xwbuf, rows, G4H, H);
        k_lstm_steps<<<1, 1024, 0, stream>>>(xwbuf, whh_bf[i] + (size_t)G4H * H, hseq,
                                             hT[i] + mp * H, cT[i] + mp * H, mp, T);
    }

    // ---- TreeLSTM, bottom-up levels ----
    k_tree_lvl0<<<(64 * 32 * 512 + 255) / 256, 256, 0, stream>>>(
        feat_old, feat_cur, tW_iou, tb_iou, tr_h, tr_hbf, tr_c);
    for (int lvl = 1; lvl <= 5; ++lvl) {
        int P = 1 << (5 - lvl);
        int R = 64 * P;
        k_tree_gather<<<(R * 512 + 255) / 256, 256, 0, stream>>>(tr_h, tr_hbf, hsum_bf, hch_bf, P);
        gemm(hsum_bf, tUiou_bf, nullptr, iou, R, 1536, H);
        gemm(hch_bf, tUf_bf, nullptr, fpre, 2 * R, 512, H);
        k_tree_update<<<(R * 512 + 255) / 256, 256, 0, stream>>>(
            feat_old, feat_cur, tW_iou, tb_iou, tW_f, tb_f, iou, fpre,
            tr_h, tr_hbf, tr_c, P);
    }

    // ---- dense head ----
    k_dense_ast<<<(32 * 512 + 255) / 256, 256, 0, stream>>>(tr_h, Wdh, bdh, h_ast);
    k_dense_am<<<1, 32, 0, stream>>>(h_ast, Wmh, bmh, h_am);
    k_mrg<<<1, 64, 0, stream>>>(hT[0], cT[0], hT[1], cT[1], h_am,
                                Wgh, bgh, Wgc, bgc, h_mrg, c_mrg);
    k_final<<<(16384 + 255) / 256, 256, 0, stream>>>(h_mrg, c_mrg, hT[2], cT[2],
                                                     Wfh, bfh, Wfc, bfc, (float*)d_out);
}
